// MoE_27848567947629
// MI455X (gfx1250) — compile-verified, matrix operand order checked by reading, not run on hardware
//
#include <hip/hip_runtime.h>
#include <math.h>

// MoE BERT layer for MI455X (gfx1250): top-1 routing + routed-only expert FFN
// via f16 WMMA (v_wmma_f32_16x16x32_f16) with f32 accumulation.
// Round 2: 2M x 4N register blocking per wave -> 8 WMMAs per 12 b128 loads.

#define H_DIM 1024
#define I_DIM 4096
#define E_NUM 4

typedef __attribute__((ext_vector_type(16))) _Float16 v16h;
typedef __attribute__((ext_vector_type(8)))  _Float16 v8h;
typedef __attribute__((ext_vector_type(8)))  float    v8f;
typedef __attribute__((ext_vector_type(4)))  float    v4f;

union HalfPack {
    v16h v;
    struct { v8h lo, hi; } p;
};

// ---------------------------------------------------------------------------
// 0. zero counters
// ---------------------------------------------------------------------------
__global__ void prep_kernel(int* __restrict__ counts, int* __restrict__ cursors) {
    if (threadIdx.x < E_NUM) {
        counts[threadIdx.x]  = 0;
        cursors[threadIdx.x] = 0;
    }
}

// ---------------------------------------------------------------------------
// 1. router: logits = x @ W_router, argmax -> expert_idx, histogram counts.
// ---------------------------------------------------------------------------
__global__ void router_kernel(const float* __restrict__ x,
                              const float* __restrict__ Wr,   // [H][E]
                              int* __restrict__ expert_idx,
                              int* __restrict__ counts, int T) {
    const int wave = (blockIdx.x * blockDim.x + threadIdx.x) >> 5;
    const int lane = threadIdx.x & 31;
    if (wave >= T) return;

    const float* xr = x + (size_t)wave * H_DIM;
    float a0 = 0.f, a1 = 0.f, a2 = 0.f, a3 = 0.f;
    for (int h = lane; h < H_DIM; h += 32) {
        float xv = xr[h];
        v4f w = *(const v4f*)(Wr + h * E_NUM);
        a0 += xv * w[0]; a1 += xv * w[1]; a2 += xv * w[2]; a3 += xv * w[3];
    }
    for (int off = 16; off; off >>= 1) {
        a0 += __shfl_xor(a0, off, 32);
        a1 += __shfl_xor(a1, off, 32);
        a2 += __shfl_xor(a2, off, 32);
        a3 += __shfl_xor(a3, off, 32);
    }
    if (lane == 0) {
        float best = a0; int bi = 0;
        if (a1 > best) { best = a1; bi = 1; }   // strict > keeps first-max (jnp.argmax)
        if (a2 > best) { best = a2; bi = 2; }
        if (a3 > best) { best = a3; bi = 3; }
        expert_idx[wave] = bi;
        atomicAdd(&counts[bi], 1);
    }
}

// ---------------------------------------------------------------------------
// 2a. exclusive scan over E=4 counts
// ---------------------------------------------------------------------------
__global__ void scan_kernel(const int* __restrict__ counts, int* __restrict__ offsets) {
    int acc = 0;
    for (int e = 0; e < E_NUM; ++e) { offsets[e] = acc; acc += counts[e]; }
}

// 2b. scatter token ids into per-expert contiguous lists
__global__ void scatter_kernel(const int* __restrict__ expert_idx,
                               const int* __restrict__ offsets,
                               int* __restrict__ cursors,
                               int* __restrict__ sorted, int T) {
    int t = blockIdx.x * blockDim.x + threadIdx.x;
    if (t >= T) return;
    int e = expert_idx[t];
    int pos = atomicAdd(&cursors[e], 1);
    sorted[offsets[e] + pos] = t;
}

// ---------------------------------------------------------------------------
// 3a. cast x -> f16
// ---------------------------------------------------------------------------
__global__ void cvt_x_kernel(const float* __restrict__ src,
                             _Float16* __restrict__ dst, size_t n) {
    size_t i = (size_t)blockIdx.x * blockDim.x + threadIdx.x;
    if (i < n) dst[i] = (_Float16)src[i];
}

// 3b. tiled transpose + cast: src[R][C] f32 -> dst[C][R] f16 (per z-slice).
__global__ void transpose_cvt_kernel(const float* __restrict__ src,
                                     _Float16* __restrict__ dst, int R, int C) {
    __shared__ float tile[32][33];
    const size_t slice = (size_t)R * C;
    const float* s = src + (size_t)blockIdx.z * slice;
    _Float16*    d = dst + (size_t)blockIdx.z * slice;
    const int c0 = blockIdx.x * 32, r0 = blockIdx.y * 32;
    for (int j = threadIdx.y; j < 32; j += blockDim.y)
        tile[j][threadIdx.x] = s[(size_t)(r0 + j) * C + c0 + threadIdx.x];
    __syncthreads();
    for (int j = threadIdx.y; j < 32; j += blockDim.y)
        d[(size_t)(c0 + j) * R + r0 + threadIdx.x] = (_Float16)tile[threadIdx.x][j];
}

__device__ __forceinline__ float gelu_exact(float v) {
    return 0.5f * v * (1.0f + erff(v * 0.70710678118654752f));
}

// ---------------------------------------------------------------------------
// 4. GEMM1 + bias + exact GeLU, routed-only.
//    Block: 8 waves, 32 tokens x 512 I-columns. Wave: 32x64 patch = 2Mx4N
//    WMMA tiles; per K-step (32): 12 b128 loads feed 8 v_wmma.
//    Per-lane A/B layout per ISA 7.12.2: lanes<16 K{0-7,16-23}, lanes>=16
//    K{8-15,24-31}, each as two b128 loads.
// ---------------------------------------------------------------------------
__global__ void __launch_bounds__(256)
gemm1_gelu_kernel(const _Float16* __restrict__ x16,   // [T][H]
                  const _Float16* __restrict__ W1t,   // [E][I][H]
                  const float*    __restrict__ b1,    // [E][I]
                  const int* __restrict__ sorted,
                  const int* __restrict__ counts,
                  const int* __restrict__ offsets,
                  _Float16* __restrict__ h16,         // [T][I]
                  int T) {
    const int e     = blockIdx.z;
    const int cnt   = counts[e];
    const int mBase = blockIdx.y * 32;
    if (mBase >= cnt) return;                 // block-uniform exit
    const int off = offsets[e];

    __shared__ int s_tok[32];
    if (threadIdx.x < 32) {
        int m = mBase + (int)threadIdx.x;
        s_tok[threadIdx.x] = sorted[off + min(m, cnt - 1)];   // clamp tail rows
    }
    __syncthreads();

    const int lane = threadIdx.x & 31;
    const int wave = threadIdx.x >> 5;
    const int half = lane >> 4;
    const int l16  = lane & 15;
    const int n0   = blockIdx.x * 512 + wave * 64;   // wave's 64-col patch

    const _Float16* pa0 = x16 + (size_t)s_tok[l16]      * H_DIM + half * 8;
    const _Float16* pa1 = x16 + (size_t)s_tok[16 + l16] * H_DIM + half * 8;
    const _Float16* pb0 = W1t + ((size_t)e * I_DIM + n0      + l16) * H_DIM + half * 8;
    const _Float16* pb1 = W1t + ((size_t)e * I_DIM + n0 + 16 + l16) * H_DIM + half * 8;
    const _Float16* pb2 = W1t + ((size_t)e * I_DIM + n0 + 32 + l16) * H_DIM + half * 8;
    const _Float16* pb3 = W1t + ((size_t)e * I_DIM + n0 + 48 + l16) * H_DIM + half * 8;

    v8f c00 = {}, c01 = {}, c02 = {}, c03 = {};
    v8f c10 = {}, c11 = {}, c12 = {}, c13 = {};

    for (int k0 = 0; k0 < H_DIM; k0 += 32) {
        HalfPack a0, a1, b0, b1v, b2, b3;
        a0.p.lo  = *(const v8h*)(pa0 + k0);  a0.p.hi  = *(const v8h*)(pa0 + k0 + 16);
        a1.p.lo  = *(const v8h*)(pa1 + k0);  a1.p.hi  = *(const v8h*)(pa1 + k0 + 16);
        b0.p.lo  = *(const v8h*)(pb0 + k0);  b0.p.hi  = *(const v8h*)(pb0 + k0 + 16);
        b1v.p.lo = *(const v8h*)(pb1 + k0);  b1v.p.hi = *(const v8h*)(pb1 + k0 + 16);
        b2.p.lo  = *(const v8h*)(pb2 + k0);  b2.p.hi  = *(const v8h*)(pb2 + k0 + 16);
        b3.p.lo  = *(const v8h*)(pb3 + k0);  b3.p.hi  = *(const v8h*)(pb3 + k0 + 16);
        c00 = __builtin_amdgcn_wmma_f32_16x16x32_f16(false, a0.v, false, b0.v,  (short)0, c00, false, false);
        c01 = __builtin_amdgcn_wmma_f32_16x16x32_f16(false, a0.v, false, b1v.v, (short)0, c01, false, false);
        c02 = __builtin_amdgcn_wmma_f32_16x16x32_f16(false, a0.v, false, b2.v,  (short)0, c02, false, false);
        c03 = __builtin_amdgcn_wmma_f32_16x16x32_f16(false, a0.v, false, b3.v,  (short)0, c03, false, false);
        c10 = __builtin_amdgcn_wmma_f32_16x16x32_f16(false, a1.v, false, b0.v,  (short)0, c10, false, false);
        c11 = __builtin_amdgcn_wmma_f32_16x16x32_f16(false, a1.v, false, b1v.v, (short)0, c11, false, false);
        c12 = __builtin_amdgcn_wmma_f32_16x16x32_f16(false, a1.v, false, b2.v,  (short)0, c12, false, false);
        c13 = __builtin_amdgcn_wmma_f32_16x16x32_f16(false, a1.v, false, b3.v,  (short)0, c13, false, false);
    }

    const float bias0 = b1[(size_t)e * I_DIM + n0      + l16];
    const float bias1 = b1[(size_t)e * I_DIM + n0 + 16 + l16];
    const float bias2 = b1[(size_t)e * I_DIM + n0 + 32 + l16];
    const float bias3 = b1[(size_t)e * I_DIM + n0 + 48 + l16];

    #pragma unroll
    for (int i = 0; i < 2; ++i) {
        const v8f* ci[4] = { i ? &c10 : &c00, i ? &c11 : &c01,
                             i ? &c12 : &c02, i ? &c13 : &c03 };
        const float bj[4] = { bias0, bias1, bias2, bias3 };
        #pragma unroll
        for (int r = 0; r < 8; ++r) {
            int mloc = i * 16 + r + 8 * half;       // C layout: M = r + 8*(lane/16)
            if (mBase + mloc < cnt) {
                const size_t rowOff = (size_t)s_tok[mloc] * I_DIM + n0 + l16;
                #pragma unroll
                for (int j = 0; j < 4; ++j) {
                    float v = (*ci[j])[r] + bj[j];
                    h16[rowOff + j * 16] = (_Float16)gelu_exact(v);
                }
            }
        }
    }
}

// ---------------------------------------------------------------------------
// 5. GEMM2 + bo + residual -> y (pre-LayerNorm, f32). K = I = 4096.
//    Same 2Mx4N blocking; rows are direct token ids.
// ---------------------------------------------------------------------------
__global__ void __launch_bounds__(256)
gemm2_kernel(const _Float16* __restrict__ h16,  // [T][I]
             const _Float16* __restrict__ Wot,  // [H][I]
             const float*    __restrict__ bo,   // [H]
             const float*    __restrict__ x,    // [T][H] residual
             float* __restrict__ y,             // [T][H]
             int T) {
    const int mBase = blockIdx.y * 32;
    const int lane  = threadIdx.x & 31;
    const int wave  = threadIdx.x >> 5;
    const int half  = lane >> 4;
    const int l16   = lane & 15;
    const int n0    = blockIdx.x * 512 + wave * 64;

    const int rowA0 = min(mBase + l16,      T - 1);
    const int rowA1 = min(mBase + 16 + l16, T - 1);
    const _Float16* pa0 = h16 + (size_t)rowA0 * I_DIM + half * 8;
    const _Float16* pa1 = h16 + (size_t)rowA1 * I_DIM + half * 8;
    const _Float16* pb0 = Wot + (size_t)(n0      + l16) * I_DIM + half * 8;
    const _Float16* pb1 = Wot + (size_t)(n0 + 16 + l16) * I_DIM + half * 8;
    const _Float16* pb2 = Wot + (size_t)(n0 + 32 + l16) * I_DIM + half * 8;
    const _Float16* pb3 = Wot + (size_t)(n0 + 48 + l16) * I_DIM + half * 8;

    v8f c00 = {}, c01 = {}, c02 = {}, c03 = {};
    v8f c10 = {}, c11 = {}, c12 = {}, c13 = {};

    for (int k0 = 0; k0 < I_DIM; k0 += 32) {
        HalfPack a0, a1, b0, b1v, b2, b3;
        a0.p.lo  = *(const v8h*)(pa0 + k0);  a0.p.hi  = *(const v8h*)(pa0 + k0 + 16);
        a1.p.lo  = *(const v8h*)(pa1 + k0);  a1.p.hi  = *(const v8h*)(pa1 + k0 + 16);
        b0.p.lo  = *(const v8h*)(pb0 + k0);  b0.p.hi  = *(const v8h*)(pb0 + k0 + 16);
        b1v.p.lo = *(const v8h*)(pb1 + k0);  b1v.p.hi = *(const v8h*)(pb1 + k0 + 16);
        b2.p.lo  = *(const v8h*)(pb2 + k0);  b2.p.hi  = *(const v8h*)(pb2 + k0 + 16);
        b3.p.lo  = *(const v8h*)(pb3 + k0);  b3.p.hi  = *(const v8h*)(pb3 + k0 + 16);
        c00 = __builtin_amdgcn_wmma_f32_16x16x32_f16(false, a0.v, false, b0.v,  (short)0, c00, false, false);
        c01 = __builtin_amdgcn_wmma_f32_16x16x32_f16(false, a0.v, false, b1v.v, (short)0, c01, false, false);
        c02 = __builtin_amdgcn_wmma_f32_16x16x32_f16(false, a0.v, false, b2.v,  (short)0, c02, false, false);
        c03 = __builtin_amdgcn_wmma_f32_16x16x32_f16(false, a0.v, false, b3.v,  (short)0, c03, false, false);
        c10 = __builtin_amdgcn_wmma_f32_16x16x32_f16(false, a1.v, false, b0.v,  (short)0, c10, false, false);
        c11 = __builtin_amdgcn_wmma_f32_16x16x32_f16(false, a1.v, false, b1v.v, (short)0, c11, false, false);
        c12 = __builtin_amdgcn_wmma_f32_16x16x32_f16(false, a1.v, false, b2.v,  (short)0, c12, false, false);
        c13 = __builtin_amdgcn_wmma_f32_16x16x32_f16(false, a1.v, false, b3.v,  (short)0, c13, false, false);
    }

    const float bias0 = bo[n0      + l16];
    const float bias1 = bo[n0 + 16 + l16];
    const float bias2 = bo[n0 + 32 + l16];
    const float bias3 = bo[n0 + 48 + l16];

    #pragma unroll
    for (int i = 0; i < 2; ++i) {
        const v8f* ci[4] = { i ? &c10 : &c00, i ? &c11 : &c01,
                             i ? &c12 : &c02, i ? &c13 : &c03 };
        const float bj[4] = { bias0, bias1, bias2, bias3 };
        #pragma unroll
        for (int r = 0; r < 8; ++r) {
            int t = mBase + i * 16 + r + 8 * half;
            if (t < T) {
                const size_t rowOff = (size_t)t * H_DIM + n0 + l16;
                #pragma unroll
                for (int j = 0; j < 4; ++j)
                    y[rowOff + j * 16] = (*ci[j])[r] + bj[j] + x[rowOff + j * 16];
            }
        }
    }
}

// ---------------------------------------------------------------------------
// 6. LayerNorm per token (block of 256 = 8 waves, H = 1024)
// ---------------------------------------------------------------------------
__global__ void layernorm_kernel(const float* __restrict__ y,
                                 const float* __restrict__ gamma,
                                 const float* __restrict__ beta,
                                 float* __restrict__ out, int T) {
    const int t = blockIdx.x;
    const float* yr = y + (size_t)t * H_DIM;
    __shared__ float s_sum[8], s_sq[8];

    float sum = 0.f, sq = 0.f;
    for (int h = threadIdx.x; h < H_DIM; h += blockDim.x) {
        float v = yr[h]; sum += v; sq += v * v;
    }
    for (int o = 16; o; o >>= 1) {
        sum += __shfl_xor(sum, o, 32);
        sq  += __shfl_xor(sq,  o, 32);
    }
    const int wave = threadIdx.x >> 5, lane = threadIdx.x & 31;
    if (lane == 0) { s_sum[wave] = sum; s_sq[wave] = sq; }
    __syncthreads();
    float ts = 0.f, tq = 0.f;
    #pragma unroll
    for (int i = 0; i < 8; ++i) { ts += s_sum[i]; tq += s_sq[i]; }

    const float mu  = ts * (1.0f / H_DIM);
    const float var = tq * (1.0f / H_DIM) - mu * mu;
    const float rs  = rsqrtf(var + 1e-12f);
    for (int h = threadIdx.x; h < H_DIM; h += blockDim.x)
        out[(size_t)t * H_DIM + h] = (yr[h] - mu) * rs * gamma[h] + beta[h];
}

// ---------------------------------------------------------------------------
// launch
// ---------------------------------------------------------------------------
extern "C" void kernel_launch(void* const* d_in, const int* in_sizes, int n_in,
                              void* d_out, int out_size, void* d_ws, size_t ws_size,
                              hipStream_t stream) {
    const float* x   = (const float*)d_in[0];   // [T][H]
    const float* Wr  = (const float*)d_in[1];   // [H][E]
    const float* W1  = (const float*)d_in[2];   // [E][H][I]
    const float* b1  = (const float*)d_in[3];   // [E][I]
    const float* Wo  = (const float*)d_in[4];   // [I][H]
    const float* bo  = (const float*)d_in[5];   // [H]
    const float* g   = (const float*)d_in[6];   // [H]
    const float* bt  = (const float*)d_in[7];   // [H]
    float* out = (float*)d_out;

    const int T = in_sizes[0] / H_DIM;          // 2048 for the reference shapes

    // carve workspace (aligned 256B): ~71 MB total
    char* p = (char*)d_ws;
    auto carve = [&](size_t bytes) -> char* {
        char* r = p; p += (bytes + 255) & ~(size_t)255; return r;
    };
    int* expert_idx = (int*)carve((size_t)T * 4);
    int* counts     = (int*)carve(64);
    int* offsets    = (int*)carve(64);
    int* cursors    = (int*)carve(64);
    int* sorted     = (int*)carve((size_t)T * 4);
    _Float16* x16   = (_Float16*)carve((size_t)T * H_DIM * 2);
    _Float16* W1t   = (_Float16*)carve((size_t)E_NUM * H_DIM * I_DIM * 2);
    _Float16* Wot   = (_Float16*)carve((size_t)H_DIM * I_DIM * 2);
    _Float16* h16   = (_Float16*)carve((size_t)T * I_DIM * 2);
    float*    ybuf  = (float*)carve((size_t)T * H_DIM * 4);

    // 0-2: routing
    prep_kernel<<<1, 32, 0, stream>>>(counts, cursors);
    router_kernel<<<(T * 32 + 255) / 256, 256, 0, stream>>>(x, Wr, expert_idx, counts, T);
    scan_kernel<<<1, 1, 0, stream>>>(counts, offsets);
    scatter_kernel<<<(T + 255) / 256, 256, 0, stream>>>(expert_idx, offsets, cursors, sorted, T);

    // 3: precision/layout prep
    {
        size_t n = (size_t)T * H_DIM;
        cvt_x_kernel<<<(unsigned)((n + 255) / 256), 256, 0, stream>>>(x, x16, n);
    }
    dim3 tb(32, 8);
    transpose_cvt_kernel<<<dim3(I_DIM / 32, H_DIM / 32, E_NUM), tb, 0, stream>>>(W1, W1t, H_DIM, I_DIM);
    transpose_cvt_kernel<<<dim3(H_DIM / 32, I_DIM / 32, 1),     tb, 0, stream>>>(Wo, Wot, I_DIM, H_DIM);

    // 4: expert GEMM1 + GeLU (routed-only): 32 tokens x 512 cols per block
    gemm1_gelu_kernel<<<dim3(I_DIM / 512, (T + 31) / 32, E_NUM), 256, 0, stream>>>(
        x16, W1t, b1, sorted, counts, offsets, h16, T);

    // 5: GEMM2 + bias + residual: 32 tokens x 512 cols per block
    gemm2_kernel<<<dim3(H_DIM / 512, (T + 31) / 32), 256, 0, stream>>>(
        h16, Wot, bo, x, ybuf, T);

    // 6: LayerNorm
    layernorm_kernel<<<T, 256, 0, stream>>>(ybuf, g, bt, out, T);
}
